// GATLayer_37864431681685
// MI455X (gfx1250) — compile-verified, matrix operand order checked by reading, not run on hardware
//
#include <hip/hip_runtime.h>
#include <math.h>

// GAT reference collapses: einsum('hid,nf->hnd') has mismatched contraction
// labels => Wh[h,n,d] = ws[h,d]*xs[n] with ws=colsum(W), xs=rowsum(x).
// Pipeline: rowsum(x) on the WMMA pipe (HBM-bound, compute free), tiny prep,
// ONE fused edge pass for the global per-head softmax max+sum (online
// softmax, block pair-reduction), one edge scatter pass (8 f32 atomics/edge
// into an L2-resident 3.2MB accumulator), then the elementwise finalize.

typedef __attribute__((ext_vector_type(16))) _Float16 v16h;
typedef __attribute__((ext_vector_type(8)))  float    v8f;

#define F_IN 128
#define NH   8
#define ND   8
#define TPB  256
#define EPT  8
#define NEG_SLOPE 0.2f

// ---------------- K0: zero G ----------------
__global__ void k_zero(float* __restrict__ p, long n) {
    long i = (long)blockIdx.x * blockDim.x + threadIdx.x;
    if (i < n) p[i] = 0.0f;
}

// ---------------- K1: xs[n] = sum_f x[n][f] via WMMA ----------------
// Each wave handles 16 rows; A tile = 16x32 f16 slice of x, B = ones(32x16).
// Column 0 of the 16x16 f32 C tile carries the row sums (lanes 0 and 16).
__global__ void k_rowsum_wmma(const float* __restrict__ x,
                              float* __restrict__ xs, int N) {
    const int lane = threadIdx.x & 31;
    const int wave = threadIdx.x >> 5;
    const int row0 = (blockIdx.x * 8 + wave) * 16;
    if (row0 >= N) return;                 // wave-uniform: EXEC stays all-ones
    const bool lo = lane < 16;
    int row = row0 + (lane & 15);
    if (row > N - 1) row = N - 1;          // clamp (duplicate loads, guarded store)
    const float* xr = x + (long)row * F_IN;

    v16h ones;
#pragma unroll
    for (int j = 0; j < 16; ++j) ones[j] = (_Float16)1.0f;

    v8f acc = {};
#pragma unroll
    for (int k0 = 0; k0 < F_IN; k0 += 32) {
        const int kb = k0 + (lo ? 0 : 8);  // ISA 16-bit A fragment layout
        v16h a;
#pragma unroll
        for (int j = 0; j < 8; ++j) {
            a[j]     = (_Float16)xr[kb + j];        // K = kb .. kb+7
            a[j + 8] = (_Float16)xr[kb + 16 + j];   // K = kb+16 .. kb+23
        }
        acc = __builtin_amdgcn_wmma_f32_16x16x32_f16(
                  false, a, false, ones, (short)0, acc, false, false);
    }
    // C layout: VGPR r -> (M=r, N=lane) lanes 0-15; (M=8+r, N=lane-16) lanes 16-31.
    // Each storing lane owns 8 consecutive rows => contiguous 32B in xs.
    if ((lane & 15) == 0) {
        const int rbase = row0 + (lo ? 0 : 8);
        if (rbase + 8 <= N) {
            float4 v0 = make_float4(acc[0], acc[1], acc[2], acc[3]);
            float4 v1 = make_float4(acc[4], acc[5], acc[6], acc[7]);
            *(float4*)(xs + rbase)     = v0;   // rbase is 8-aligned -> 32B aligned
            *(float4*)(xs + rbase + 4) = v1;
        } else {
#pragma unroll
            for (int r = 0; r < 8; ++r) {
                const int rr = rbase + r;
                if (rr < N) xs[rr] = acc[r];
            }
        }
    }
}

// ---------------- K2: ws[h,d], A[h], B[h] (tiny) ----------------
__global__ void k_prep(const float* __restrict__ W, const float* __restrict__ a,
                       float* __restrict__ wsum, float* __restrict__ Ah,
                       float* __restrict__ Bh) {
    __shared__ float s[NH * ND];
    const int t = threadIdx.x;             // 64 threads
    if (t < NH * ND) {
        const int h = t >> 3, d = t & 7;
        float acc = 0.0f;
        for (int i = 0; i < F_IN; ++i) acc += W[h * F_IN * ND + i * ND + d];
        s[t] = acc;
        wsum[t] = acc;
    }
    __syncthreads();
    if (t < NH) {
        float A = 0.0f, B = 0.0f;
#pragma unroll
        for (int d = 0; d < ND; ++d) {
            A += s[t * ND + d] * a[t * 16 + d];
            B += s[t * ND + d] * a[t * 16 + ND + d];
        }
        Ah[t] = A;
        Bh[t] = B;
    }
}

__device__ __forceinline__ float lrelu(float e) {
    return e > 0.0f ? e : NEG_SLOPE * e;
}

// Pairwise (max, sum) tree reduction for online softmax.
// Combine: M = max(m1,m2); S = s1*exp(m1-M) + s2*exp(m2-M), guarding m = -inf.
__device__ __forceinline__ void block_reduce_ms(float m[NH], float s[NH],
                                                float* mred, float* sred) {
    const int t = threadIdx.x;
#pragma unroll
    for (int h = 0; h < NH; ++h) { mred[h * TPB + t] = m[h]; sred[h * TPB + t] = s[h]; }
    __syncthreads();
    for (int st = TPB / 2; st > 0; st >>= 1) {
        if (t < st) {
#pragma unroll
            for (int h = 0; h < NH; ++h) {
                const float m1 = mred[h * TPB + t],      s1 = sred[h * TPB + t];
                const float m2 = mred[h * TPB + t + st], s2 = sred[h * TPB + t + st];
                const float M  = fmaxf(m1, m2);
                const float e1 = (m1 > -INFINITY) ? __expf(m1 - M) : 0.0f;
                const float e2 = (m2 > -INFINITY) ? __expf(m2 - M) : 0.0f;
                mred[h * TPB + t] = M;
                sred[h * TPB + t] = s1 * e1 + s2 * e2;
            }
        }
        __syncthreads();
    }
}

// ---------------- K3: fused per-block online softmax (max AND sum) ----------
__global__ void k_edge_maxsum(const long long* __restrict__ ei,
                              const float* __restrict__ xs,
                              const float* __restrict__ Ah, const float* __restrict__ Bh,
                              float* __restrict__ pmax, float* __restrict__ psum,
                              long E) {
    __shared__ float mred[NH * TPB];
    __shared__ float sred[NH * TPB];
    float A[NH], B[NH], mx[NH], sm[NH];
#pragma unroll
    for (int h = 0; h < NH; ++h) {
        A[h] = Ah[h]; B[h] = Bh[h]; mx[h] = -INFINITY; sm[h] = 0.0f;
    }
    const long base = ((long)blockIdx.x * TPB + threadIdx.x) * EPT;
    if (base + EPT <= E) {                 // full fast path: vectorizable loads
        long long sv[EPT], dv[EPT];
#pragma unroll
        for (int j = 0; j < EPT; ++j) sv[j] = ei[base + j];
#pragma unroll
        for (int j = 0; j < EPT; ++j) dv[j] = ei[E + base + j];
#pragma unroll
        for (int j = 0; j < EPT; ++j) {
            const float xss = xs[(int)sv[j]];
            const float xsd = xs[(int)dv[j]];
#pragma unroll
            for (int h = 0; h < NH; ++h) {
                const float e  = lrelu(xss * A[h] + xsd * B[h]);
                const float m0 = mx[h];
                const float nm = fmaxf(m0, e);
                mx[h] = nm;
                sm[h] = sm[h] * __expf(m0 - nm) + __expf(e - nm); // exp(-inf)=0 ok
            }
        }
    } else {
        for (long i = base; i < E && i < base + EPT; ++i) {
            const float xss = xs[(int)ei[i]];
            const float xsd = xs[(int)ei[E + i]];
#pragma unroll
            for (int h = 0; h < NH; ++h) {
                const float e  = lrelu(xss * A[h] + xsd * B[h]);
                const float m0 = mx[h];
                const float nm = fmaxf(m0, e);
                mx[h] = nm;
                sm[h] = sm[h] * __expf(m0 - nm) + __expf(e - nm);
            }
        }
    }
    block_reduce_ms(mx, sm, mred, sred);
    if (threadIdx.x < NH) {
        pmax[(long)blockIdx.x * NH + threadIdx.x] = mred[threadIdx.x * TPB];
        psum[(long)blockIdx.x * NH + threadIdx.x] = sred[threadIdx.x * TPB];
    }
}

// ---------------- K4: combine block pairs -> headMax[8], headInv[8] ---------
__global__ void k_combine(const float* __restrict__ pmax,
                          const float* __restrict__ psum,
                          float* __restrict__ headMax, float* __restrict__ headInv,
                          int nblk) {
    __shared__ float mred[NH * TPB];
    __shared__ float sred[NH * TPB];
    float mx[NH], sm[NH];
#pragma unroll
    for (int h = 0; h < NH; ++h) { mx[h] = -INFINITY; sm[h] = 0.0f; }
    for (int i = threadIdx.x; i < nblk; i += TPB) {
#pragma unroll
        for (int h = 0; h < NH; ++h) {
            const float m2 = pmax[(long)i * NH + h];
            const float s2 = psum[(long)i * NH + h];
            const float m0 = mx[h];
            const float M  = fmaxf(m0, m2);
            const float e1 = (m0 > -INFINITY) ? __expf(m0 - M) : 0.0f;
            const float e2 = (m2 > -INFINITY) ? __expf(m2 - M) : 0.0f;
            mx[h] = M;
            sm[h] = sm[h] * e1 + s2 * e2;
        }
    }
    block_reduce_ms(mx, sm, mred, sred);
    if (threadIdx.x < NH) {
        headMax[threadIdx.x] = mred[threadIdx.x * TPB];
        headInv[threadIdx.x] = 1.0f / sred[threadIdx.x * TPB];
    }
}

// ---------------- K5: scatter G[dst,h] += xs[src]*alpha (8 atomics/edge) ----
__global__ void k_edge_scatter(const long long* __restrict__ ei,
                               const float* __restrict__ xs,
                               const float* __restrict__ Ah, const float* __restrict__ Bh,
                               const float* __restrict__ headMax,
                               const float* __restrict__ headInv,
                               float* __restrict__ G, long E) {
    float A[NH], B[NH], M[NH], I[NH];
#pragma unroll
    for (int h = 0; h < NH; ++h) {
        A[h] = Ah[h]; B[h] = Bh[h]; M[h] = headMax[h]; I[h] = headInv[h];
    }
    const long base = ((long)blockIdx.x * TPB + threadIdx.x) * EPT;
    if (base + EPT <= E) {
        long long sv[EPT], dv[EPT];
#pragma unroll
        for (int j = 0; j < EPT; ++j) sv[j] = ei[base + j];
#pragma unroll
        for (int j = 0; j < EPT; ++j) dv[j] = ei[E + base + j];
#pragma unroll
        for (int j = 0; j < EPT; ++j) {
            const int   d   = (int)dv[j];
            const float xss = xs[(int)sv[j]];
            const float xsd = xs[d];
#pragma unroll
            for (int h = 0; h < NH; ++h) {
                const float e     = lrelu(xss * A[h] + xsd * B[h]);
                const float alpha = __expf(e - M[h]) * I[h];
                atomicAdd(&G[(long)d * NH + h], xss * alpha);
            }
        }
    } else {
        for (long i = base; i < E && i < base + EPT; ++i) {
            const int   d   = (int)ei[E + i];
            const float xss = xs[(int)ei[i]];
            const float xsd = xs[d];
#pragma unroll
            for (int h = 0; h < NH; ++h) {
                const float e     = lrelu(xss * A[h] + xsd * B[h]);
                const float alpha = __expf(e - M[h]) * I[h];
                atomicAdd(&G[(long)d * NH + h], xss * alpha);
            }
        }
    }
}

// ---------------- K6: out[n, h*8+d] = elu(ws[h,d]*G[n,h] + bias[h,d]) -------
__global__ void k_finalize(const float* __restrict__ G,
                           const float* __restrict__ wsum,
                           const float* __restrict__ bias,
                           float* __restrict__ out, long total) {
    const long i = (long)blockIdx.x * blockDim.x + threadIdx.x;
    if (i >= total) return;
    const int c = (int)(i & 63);           // h*8 + d
    const long n = i >> 6;
    const int h = c >> 3;
    const float v = wsum[c] * G[n * NH + h] + bias[c];
    out[i] = v > 0.0f ? v : expm1f(v);
}

extern "C" void kernel_launch(void* const* d_in, const int* in_sizes, int n_in,
                              void* d_out, int out_size, void* d_ws, size_t ws_size,
                              hipStream_t stream) {
    const float*     x    = (const float*)d_in[0];
    const long long* ei   = (const long long*)d_in[1];
    const float*     W    = (const float*)d_in[2];
    const float*     a    = (const float*)d_in[3];
    const float*     bias = (const float*)d_in[4];
    float*           out  = (float*)d_out;

    const int  N = in_sizes[0] / F_IN;           // 100000
    const long E = (long)in_sizes[1] / 2;        // 1600000
    const int  PBLK = (int)((E + (long)TPB * EPT - 1) / ((long)TPB * EPT));

    // workspace layout (floats)
    float* wsf     = (float*)d_ws;
    float* xs      = wsf;                         // N
    float* G       = xs + N;                      // N*8
    float* wsum    = G + (size_t)N * NH;          // 64
    float* Ah      = wsum + NH * ND;              // 8
    float* Bh      = Ah + NH;                     // 8
    float* headMax = Bh + NH;                     // 8
    float* headInv = headMax + NH;                // 8
    float* pmax    = headInv + NH;                // PBLK*8
    float* psum    = pmax + (size_t)PBLK * NH;    // PBLK*8

    const long gsize = (long)N * NH;
    k_zero<<<(int)((gsize + TPB - 1) / TPB), TPB, 0, stream>>>(G, gsize);

    k_rowsum_wmma<<<(N + 127) / 128, TPB, 0, stream>>>(x, xs, N);

    k_prep<<<1, 64, 0, stream>>>(W, a, wsum, Ah, Bh);

    k_edge_maxsum<<<PBLK, TPB, 0, stream>>>(ei, xs, Ah, Bh, pmax, psum, E);
    k_combine<<<1, TPB, 0, stream>>>(pmax, psum, headMax, headInv, PBLK);

    k_edge_scatter<<<PBLK, TPB, 0, stream>>>(ei, xs, Ah, Bh, headMax, headInv, G, E);

    const long total = (long)N * NH * ND;
    k_finalize<<<(int)((total + TPB - 1) / TPB), TPB, 0, stream>>>(G, wsum, bias, out, total);
}
